// HMLSTM_83631603188455
// MI455X (gfx1250) — matrix operand order repaired and checked by reading, not compiled, not run on hardware
//
#include <hip/hip_runtime.h>
#include <hip/hip_fp16.h>

typedef _Float16 half16 __attribute__((ext_vector_type(16)));
typedef float    float8 __attribute__((ext_vector_type(8)));

#define HM_H 256
#define HM_B 64
#define HM_T 512
#define HM_L 3
#define HM_GC 1025
#define KT_TILES 24   // 768 / 32
#define NWG 8

union ABu { half16 v; uint4 q[2]; _Float16 h[16]; };

__device__ __forceinline__ float sigf(float x) { return 1.0f / (1.0f + __expf(-x)); }

__device__ __forceinline__ void grid_barrier(unsigned* cnt, unsigned target) {
    __threadfence();
    __syncthreads();
    if (threadIdx.x == 0) {
        atomicAdd(cnt, 1u);
        while (__hip_atomic_load(cnt, __ATOMIC_RELAXED, __HIP_MEMORY_SCOPE_AGENT) < target) {
            __builtin_amdgcn_s_sleep(1);
        }
    }
    __syncthreads();
    __threadfence();
}

// One K "part" (8 K-tiles of 32): A rows from an f16 state buffer, optional row scale.
// Branch-free, fully unrolled so the scheduler can overlap loads with WMMAs.
template <bool SCALE>
__device__ __forceinline__ void gemm_part_f16(
    float8 (&acc)[4],
    const _Float16* __restrict__ arow,   // lane's row base, already + 8*hi
    const _Float16 sc,
    const _Float16* __restrict__ wpart,  // weight tiles for this part (8 kt x 64 nt x 512)
    const int hc, const int lane)
{
    #pragma unroll
    for (int kk = 0; kk < 8; ++kk) {
        if (kk + 1 < 8)
            __builtin_prefetch(wpart + (((long)(kk + 1) * 64 + hc) << 9), 0, 3);
        ABu A;
        const _Float16* hp = arow + kk * 32;
        A.q[0] = *(const uint4*)hp;
        A.q[1] = *(const uint4*)(hp + 16);
        if (SCALE) A.v = A.v * sc;
        #pragma unroll
        for (int g = 0; g < 4; ++g) {
            const _Float16* bp = wpart + (((long)kk * 64 + g * 16 + hc) << 9) + (lane << 4);
            ABu Bf;
            Bf.q[0] = *(const uint4*)bp;
            Bf.q[1] = *(const uint4*)(bp + 8);
            acc[g] = __builtin_amdgcn_wmma_f32_16x16x32_f16(
                false, A.v, false, Bf.v, (short)0, acc[g], false, false);
        }
    }
}

// Layer-0 bottom part: A rows from f32 input x (z_bottom == 1), converted inline.
__device__ __forceinline__ void gemm_part_x(
    float8 (&acc)[4],
    const float* __restrict__ xrow,      // lane's row base, already + 8*hi
    const _Float16* __restrict__ wpart,
    const int hc, const int lane)
{
    #pragma unroll
    for (int kk = 0; kk < 8; ++kk) {
        if (kk + 1 < 8)
            __builtin_prefetch(wpart + (((long)(kk + 1) * 64 + hc) << 9), 0, 3);
        const float* xp = xrow + kk * 32;
        const float4 f0 = *(const float4*)(xp + 0);
        const float4 f1 = *(const float4*)(xp + 4);
        const float4 f2 = *(const float4*)(xp + 16);
        const float4 f3 = *(const float4*)(xp + 20);
        half16 av;
        av[0]=(_Float16)f0.x; av[1]=(_Float16)f0.y; av[2]=(_Float16)f0.z; av[3]=(_Float16)f0.w;
        av[4]=(_Float16)f1.x; av[5]=(_Float16)f1.y; av[6]=(_Float16)f1.z; av[7]=(_Float16)f1.w;
        av[8]=(_Float16)f2.x; av[9]=(_Float16)f2.y; av[10]=(_Float16)f2.z; av[11]=(_Float16)f2.w;
        av[12]=(_Float16)f3.x; av[13]=(_Float16)f3.y; av[14]=(_Float16)f3.z; av[15]=(_Float16)f3.w;
        #pragma unroll
        for (int g = 0; g < 4; ++g) {
            const _Float16* bp = wpart + (((long)kk * 64 + g * 16 + hc) << 9) + (lane << 4);
            ABu Bf;
            Bf.q[0] = *(const uint4*)bp;
            Bf.q[1] = *(const uint4*)(bp + 8);
            acc[g] = __builtin_amdgcn_wmma_f32_16x16x32_f16(
                false, av, false, Bf.v, (short)0, acc[g], false, false);
        }
    }
}

// -------- Prep: pack weights into WMMA B-fragment tile layout; zero state --------
__global__ void hmlstm_prep(
    const float* __restrict__ Wb0, const float* __restrict__ Wr0, const float* __restrict__ Wt0,
    const float* __restrict__ Wb1, const float* __restrict__ Wr1, const float* __restrict__ Wt1,
    const float* __restrict__ Wb2, const float* __restrict__ Wr2, const float* __restrict__ Wt2,
    _Float16* __restrict__ wpack, float* __restrict__ wz,
    float* __restrict__ c32, _Float16* __restrict__ h16, float* __restrict__ zst,
    unsigned* __restrict__ cnt)
{
    const long stride = (long)gridDim.x * blockDim.x;
    const long tid0 = (long)blockIdx.x * blockDim.x + threadIdx.x;
    const float* Wb[3] = {Wb0, Wb1, Wb2};
    const float* Wr[3] = {Wr0, Wr1, Wr2};
    const float* Wt[3] = {Wt0, Wt1, Wt2};

    // Packed tile element order: ((l*24+kt)*64+nt)*512 + lane*16 + e
    // B-frag mapping: lane r=lane&15, hi=lane>>4 -> col n_local=r, k_local = (e&7) + 8*hi + 16*(e>>3)
    const long total_pack = 3L * KT_TILES * 64 * 512;
    for (long ti = tid0; ti < total_pack; ti += stride) {
        const int e    = (int)(ti & 15);
        const int lane = (int)((ti >> 4) & 31);
        const int nt   = (int)((ti >> 9) & 63);
        const int rest = (int)(ti >> 15);
        const int kt   = rest % KT_TILES;
        const int l    = rest / KT_TILES;
        const int k = kt * 32 + (lane >> 4) * 8 + (e & 7) + (e >> 3) * 16;   // 0..767 in [Wb;Wr;Wt]
        const int n = nt * 16 + (lane & 15);            // 0..1023
        float v;
        if (k < 256)      v = Wb[l][k * HM_GC + n];
        else if (k < 512) v = Wr[l][(k - 256) * HM_GC + n];
        else              v = (l < 2) ? Wt[l][(k - 512) * HM_GC + n] : 0.0f; // layer2 h_top==0
        wpack[ti] = (_Float16)v;
    }
    // z-tilde column (gate col 1024), f32
    for (long i = tid0; i < 3 * 768; i += stride) {
        const int l = (int)(i / 768), k = (int)(i % 768);
        float v;
        if (k < 256)      v = Wb[l][k * HM_GC + 1024];
        else if (k < 512) v = Wr[l][(k - 256) * HM_GC + 1024];
        else              v = (l < 2) ? Wt[l][(k - 512) * HM_GC + 1024] : 0.0f;
        wz[i] = v;
    }
    // zero double-buffered state
    for (long i = tid0; i < 3L * 2 * HM_B * HM_H; i += stride) { c32[i] = 0.0f; h16[i] = (_Float16)0.0f; }
    for (long i = tid0; i < 3 * 2 * HM_B; i += stride) zst[i] = 0.0f;
    if (tid0 == 0) *cnt = 0u;
}

// -------- Main persistent kernel: 8 WGs x 8 waves = 64 waves = 4 m-tiles x 16 h-col-tiles --------
__global__ void __launch_bounds__(256, 1) hmlstm_main(
    const float* __restrict__ x,
    const float* __restrict__ b0, const float* __restrict__ b1, const float* __restrict__ b2,
    const _Float16* __restrict__ wpack, const float* __restrict__ wz,
    float* __restrict__ c32, _Float16* __restrict__ h16, float* __restrict__ zst,
    unsigned* cnt, float* __restrict__ out)
{
    const int lane = threadIdx.x & 31;
    const int w  = blockIdx.x * 8 + (threadIdx.x >> 5); // 0..63
    const int mt = w >> 4;                              // batch tile 0..3
    const int hc = w & 15;                              // h-column tile 0..15
    const int r  = lane & 15;
    const int hi = lane >> 4;
    float* out_h = out;
    float* out_z = out + (long)HM_B * HM_T * (3 * HM_H);
    unsigned bar = 0;
    const float8 zero8 = {0.f, 0.f, 0.f, 0.f, 0.f, 0.f, 0.f, 0.f};
    const long PART_W = 8L * 64 * 512;                  // weight halves per K-part

    for (int t = 0; t < HM_T; ++t) {
        const int par = t & 1, prv = par ^ 1;
        for (int l = 0; l < HM_L; ++l) {
            const float* bias = (l == 0) ? b0 : (l == 1) ? b1 : b2;
            const int brow = mt * 16 + r;               // A-fragment row for this lane
            const float zb_s = (l == 0) ? 1.0f : zst[((l - 1) * 2 + par) * HM_B + brow];
            const float zo_s = zst[(l * 2 + prv) * HM_B + brow];

            float8 acc[4];
            #pragma unroll
            for (int g = 0; g < 4; ++g) acc[g] = zero8;

            const _Float16* wl = wpack + (long)l * KT_TILES * 64 * 512;
            // part 0: z_bottom * h_bottom  (layer 0: x, z_bottom == 1)
            if (l == 0) {
                gemm_part_x(acc, x + ((long)brow * HM_T + t) * 256 + 8 * hi, wl, hc, lane);
            } else {
                gemm_part_f16<true>(acc,
                    h16 + ((((l - 1) * 2 + par) * HM_B + brow) << 8) + 8 * hi,
                    (_Float16)zb_s, wl, hc, lane);
            }
            // part 1: h (previous own state), unscaled
            gemm_part_f16<false>(acc,
                h16 + (((l * 2 + prv) * HM_B + brow) << 8) + 8 * hi,
                (_Float16)1.0f, wl + PART_W, hc, lane);
            // part 2: z * h_top (layers 0,1 only; layer 2 h_top == 0)
            if (l < 2) {
                gemm_part_f16<true>(acc,
                    h16 + ((((l + 1) * 2 + prv) * HM_B + brow) << 8) + 8 * hi,
                    (_Float16)zo_s, wl + 2 * PART_W, hc, lane);
            }

            // bias (same column for all 8 rows a lane holds)
            #pragma unroll
            for (int g = 0; g < 4; ++g) {
                const float bv = bias[g * 256 + hc * 16 + r];
                #pragma unroll
                for (int v = 0; v < 8; ++v) acc[g][v] += bv;
            }

            // z-tilde column: 4 hc==0 waves, vectorized per-row dot; halves combined via shfl
            if (hc == 0) {
                const float* wzl = wz + l * 768;
                float s = 0.f;
                const int ktot = (l == 2) ? 512 : 768;
                const int half = ktot >> 1;
                for (int kb = hi * half; kb < (hi + 1) * half; kb += 8) {
                    const int part = kb >> 8, kk = kb & 255;
                    float av[8];
                    if (part == 0 && l == 0) {
                        const float* xp = x + ((long)brow * HM_T + t) * 256 + kk;
                        const float4 g0 = *(const float4*)(xp);
                        const float4 g1 = *(const float4*)(xp + 4);
                        av[0]=g0.x; av[1]=g0.y; av[2]=g0.z; av[3]=g0.w;
                        av[4]=g1.x; av[5]=g1.y; av[6]=g1.z; av[7]=g1.w;
                    } else {
                        const _Float16* hp; float sc;
                        if (part == 0)      { hp = h16 + ((((l - 1) * 2 + par) * HM_B + brow) << 8); sc = zb_s; }
                        else if (part == 1) { hp = h16 + (((l * 2 + prv) * HM_B + brow) << 8); sc = 1.0f; }
                        else                { hp = h16 + ((((l + 1) * 2 + prv) * HM_B + brow) << 8); sc = zo_s; }
                        union { uint4 q; _Float16 hh[8]; } u;
                        u.q = *(const uint4*)(hp + kk);
                        #pragma unroll
                        for (int e = 0; e < 8; ++e) av[e] = sc * (float)u.hh[e];
                    }
                    const float4 w0 = *(const float4*)(wzl + kb);
                    const float4 w1 = *(const float4*)(wzl + kb + 4);
                    s += av[0]*w0.x + av[1]*w0.y + av[2]*w0.z + av[3]*w0.w
                       + av[4]*w1.x + av[5]*w1.y + av[6]*w1.z + av[7]*w1.w;
                }
                s += __shfl_down(s, 16, 32);
                if (hi == 0) {
                    s += bias[1024];
                    const float zh = fminf(fmaxf((s + 1.f) * 0.5f, 0.f), 1.f);
                    const float zn = rintf(zh);   // round-half-to-even == jnp.round
                    zst[(l * 2 + par) * HM_B + brow] = zn;
                    out_z[((long)brow * HM_T + t) * 3 + l] = zn;
                }
            }

            // elementwise cell update: lane holds col j, rows mt*16+8*hi+v (C/D layout)
            const int j = hc * 16 + r;
            #pragma unroll
            for (int v = 0; v < 8; ++v) {
                const int b = mt * 16 + 8 * hi + v;
                const float zo = zst[(l * 2 + prv) * HM_B + b];
                const float zb = (l == 0) ? 1.0f : zst[((l - 1) * 2 + par) * HM_B + b];
                const float fg = sigf(acc[0][v]);
                const float ig = sigf(acc[1][v]);
                const float og = sigf(acc[2][v]);
                const float gg = tanhf(acc[3][v]);
                const long ip = (((long)(l * 2 + prv) * HM_B + b) << 8) + j;
                const long ic = (((long)(l * 2 + par) * HM_B + b) << 8) + j;
                const float c = c32[ip];
                const float iggg = ig * gg;
                const float cn = zo * iggg + (1.f - zo) * (zb * (fg * c + iggg) + (1.f - zb) * c);
                const float cp = (1.f - zo) * (1.f - zb);
                const float hold = (float)h16[ip];
                const float hn = cp * hold + (1.f - cp) * (og * tanhf(cn));
                c32[ic] = cn;
                h16[ic] = (_Float16)hn;
                out_h[((long)b * HM_T + t) * (3 * HM_H) + l * 256 + j] = hn;
            }
            grid_barrier(cnt, (++bar) * NWG);
        }
    }
}

extern "C" void kernel_launch(void* const* d_in, const int* in_sizes, int n_in,
                              void* d_out, int out_size, void* d_ws, size_t ws_size,
                              hipStream_t stream) {
    (void)in_sizes; (void)n_in; (void)out_size; (void)ws_size;
    const float* x = (const float*)d_in[0];
    const float* Wb[3] = {(const float*)d_in[1], (const float*)d_in[5], (const float*)d_in[9]};
    const float* Wr[3] = {(const float*)d_in[2], (const float*)d_in[6], (const float*)d_in[10]};
    const float* Wt[3] = {(const float*)d_in[3], (const float*)d_in[7], (const float*)d_in[11]};
    const float* bb[3] = {(const float*)d_in[4], (const float*)d_in[8], (const float*)d_in[12]};

    char* p = (char*)d_ws;
    auto take = [&](size_t bytes) { char* q = p; p += (bytes + 255) & ~size_t(255); return q; };
    unsigned* cnt   = (unsigned*)take(sizeof(unsigned));
    float* zst      = (float*)take(3 * 2 * 64 * sizeof(float));
    float* wz       = (float*)take(3 * 768 * sizeof(float));
    float* c32      = (float*)take(3L * 2 * 64 * 256 * sizeof(float));
    _Float16* h16   = (_Float16*)take(3L * 2 * 64 * 256 * sizeof(_Float16));
    _Float16* wpack = (_Float16*)take(3L * KT_TILES * 64 * 512 * sizeof(_Float16));

    hmlstm_prep<<<dim3(512), dim3(256), 0, stream>>>(
        Wb[0], Wr[0], Wt[0], Wb[1], Wr[1], Wt[1], Wb[2], Wr[2], Wt[2],
        wpack, wz, c32, h16, zst, cnt);
    hmlstm_main<<<dim3(NWG), dim3(256), 0, stream>>>(
        x, bb[0], bb[1], bb[2], wpack, wz, c32, h16, zst, cnt, (float*)d_out);
}
